// DecayingBuffer_74586402063014
// MI455X (gfx1250) — matrix operand order repaired and checked by reading, not compiled, not run on hardware
//
#include <hip/hip_runtime.h>
#include <cstdint>
#include <cmath>

#define T_TOK 8192
#define DDIM  256
#define NSLOT 4096
#define SCALE 0.0625f   /* 1/sqrt(256) */

typedef __attribute__((ext_vector_type(16))) __bf16 v16bf;
typedef __attribute__((ext_vector_type(8)))  float  v8f;
typedef unsigned int u32x4 __attribute__((ext_vector_type(4)));
typedef int          i32x4 __attribute__((ext_vector_type(4)));
typedef int          i32x8 __attribute__((ext_vector_type(8)));

#if __has_builtin(__builtin_amdgcn_tensor_load_to_lds)
#define HAVE_TDM 1
#else
#define HAVE_TDM 0
#endif

union BF16Frag { v16bf v; uint32_t u[8]; };

__device__ __forceinline__ uint16_t f2bf(float f) {
  uint32_t u = __float_as_uint(f);
  if ((u & 0x7fffffffu) > 0x7f800000u) return (uint16_t)((u >> 16) | 0x0040u); // NaN
  uint32_t r = u + 0x7fffu + ((u >> 16) & 1u);  // round-to-nearest-even
  return (uint16_t)(r >> 16);
}

// A operand: row-major [M,K] bf16, K contiguous. 16x32 tile at (row0, k0).
__device__ __forceinline__ void load_a(BF16Frag& f, const uint16_t* A, int ldk,
                                       int row0, int k0, int lane) {
  const int m = lane & 15, half = lane >> 4;
  const uint32_t* p = (const uint32_t*)(A + (size_t)(row0 + m) * ldk);
#pragma unroll
  for (int e = 0; e < 8; ++e) {
    int k = k0 + ((e >= 4) ? 16 : 0) + (e & 3) * 2 + half * 8;
    f.u[e] = p[k >> 1];
  }
}

// B operand given as Bt: [Ncols, K] bf16, K contiguous. 32x16 (KxN) tile.
__device__ __forceinline__ void load_b(BF16Frag& f, const uint16_t* Bt, int ldk,
                                       int ncol0, int k0, int lane) {
  const int n = lane & 15, half = lane >> 4;
  const uint32_t* p = (const uint32_t*)(Bt + (size_t)(ncol0 + n) * ldk);
#pragma unroll
  for (int e = 0; e < 8; ++e) {
    int k = k0 + ((e >= 4) ? 16 : 0) + (e & 3) * 2 + half * 8;
    f.u[e] = p[k >> 1];
  }
}

__device__ __forceinline__ v8f wmma_bf16(const BF16Frag& a, const BF16Frag& b, v8f c) {
  return __builtin_amdgcn_wmma_f32_16x16x32_bf16(false, a.v, false, b.v,
                                                 (short)0, c, false, false);
}

// ---- CDNA5 async load to LDS (ASYNCcnt path), inline asm per ISA 15.18.3 ----
__device__ __forceinline__ void async_copy_b128(void* ldsp, const void* g) {
  uint32_t loff = (uint32_t)(uintptr_t)ldsp;   // low 32 bits of generic addr = LDS offset
  asm volatile("global_load_async_to_lds_b128 %0, %1, off"
               :: "v"(loff), "v"(g) : "memory");
}
#define ASYNC_WAIT(n) asm volatile("s_wait_asynccnt %0" :: "n"(n) : "memory")

// ---- CDNA5 Tensor Data Mover: 2-D bf16 tile, global -> LDS (TENSORcnt path) ----
// tile = tensor: elems_x per row, elems_y rows, row stride in elements.
__device__ __forceinline__ void tdm_load_2d(void* ldsp, const void* gaddr,
                                            uint32_t elems_x, uint32_t elems_y,
                                            uint32_t stride_elems) {
#if HAVE_TDM
  uint64_t ga = (uint64_t)(uintptr_t)gaddr;
  uint32_t lds_off = (uint32_t)(uintptr_t)ldsp;
  u32x4 g0;
  g0[0] = 1u;                                            // count=1, user descriptor
  g0[1] = lds_off;                                       // lds_addr (bytes)
  g0[2] = (uint32_t)(ga & 0xffffffffu);                  // global_addr[31:0]
  g0[3] = (uint32_t)((ga >> 32) & 0x01ffffffu) | (2u << 30); // addr[56:32] | type=2
  i32x8 g1;
  g1[0] = (int)(1u << 16);                               // data_size=1 -> 2 bytes/elem
  g1[1] = (int)((elems_x & 0xffffu) << 16);              // tensor_dim0[15:0]
  g1[2] = (int)(((elems_x >> 16) & 0xffffu) |
                ((elems_y & 0xffffu) << 16));            // dim0[31:16] | dim1[15:0]
  g1[3] = (int)(((elems_y >> 16) & 0xffffu) |
                ((elems_x & 0xffffu) << 16));            // dim1[31:16] | tile_dim0
  g1[4] = (int)(elems_y & 0xffffu);                      // tile_dim1 (tile_dim2=0)
  g1[5] = (int)stride_elems;                             // tensor_dim0_stride[31:0]
  g1[6] = 0;
  g1[7] = 0;
  i32x4 z4 = {0, 0, 0, 0};
#if defined(__clang_major__) && (__clang_major__ >= 23)
  i32x8 z8 = {0, 0, 0, 0, 0, 0, 0, 0};
  __builtin_amdgcn_tensor_load_to_lds(g0, g1, z4, z4, z8, 0);
#else
  __builtin_amdgcn_tensor_load_to_lds(g0, g1, z4, z4, 0);
#endif
#endif
}

// ---------------------------------------------------------------- converts
__global__ void cvt_kernel(const float* __restrict__ in, uint16_t* __restrict__ out, int n) {
  int i = blockIdx.x * blockDim.x + threadIdx.x;
  if (i < n) out[i] = f2bf(in[i]);
}

__global__ void initbuf_kernel(const float* __restrict__ k0, const float* __restrict__ v0,
                               float* __restrict__ kb, float* __restrict__ vb, int n) {
  int i = blockIdx.x * blockDim.x + threadIdx.x;
  if (i < n) { kb[i] = k0[i]; vb[i] = v0[i]; }
}

__global__ void convert2_kernel(const float* __restrict__ kb, const float* __restrict__ vb,
                                const float* __restrict__ act,
                                uint16_t* __restrict__ kbbf, uint16_t* __restrict__ vbtbf,
                                float* __restrict__ bias) {
  int i = blockIdx.x * blockDim.x + threadIdx.x;
  if (i < NSLOT * DDIM) {
    kbbf[i] = f2bf(kb[i]);
    int n = i / DDIM, d = i % DDIM;
    vbtbf[(size_t)d * NSLOT + n] = f2bf(vb[i]);
  }
  if (i < NSLOT) {
    float a = act[i];
    bias[i] = (a < 0.01f) ? -1e30f : logf(fmaxf(a, 1e-8f));
  }
}

// ---------------------------------------------------------------- phase 1: projections
// W (256x256 bf16, 128 KB) is staged once per block into LDS by the Tensor Data Mover.
__global__ __launch_bounds__(256)
void proj_gemm_kernel(const uint16_t* __restrict__ A,   // [T,256] bf16
                      const uint16_t* __restrict__ Bt,  // [256,256] bf16 (W row-major = Bt)
                      const float* __restrict__ bias,   // [256]
                      float* __restrict__ outf,         // [T,256] or null
                      uint16_t* __restrict__ outbf) {   // [T,256] or null
  __shared__ uint16_t wlds[DDIM * DDIM];                // 128 KB
  const int tid = threadIdx.x;
  const int lane = tid & 31;
  const int wave = tid >> 5;
  const int rowBase = blockIdx.x * 128 + wave * 16;
  const int m = lane & 15, half = lane >> 4;

#if HAVE_TDM
  if (wave == 0) tdm_load_2d(wlds, Bt, DDIM, DDIM, DDIM);
  __builtin_amdgcn_s_wait_tensorcnt(0);
#else
  {
    const uint32_t* g = (const uint32_t*)Bt;
    uint32_t* l = (uint32_t*)wlds;
    for (int i = tid; i < DDIM * DDIM / 2; i += 256) l[i] = g[i];
  }
#endif
  __syncthreads();

  BF16Frag qa[8];
#pragma unroll
  for (int i = 0; i < 8; ++i) load_a(qa[i], A, DDIM, rowBase, i * 32, lane);

  for (int nb = 0; nb < DDIM; nb += 16) {
    v8f acc = {0.f, 0.f, 0.f, 0.f, 0.f, 0.f, 0.f, 0.f};
#pragma unroll
    for (int i = 0; i < 8; ++i) {
      BF16Frag b;
      load_b(b, wlds, DDIM, nb, i * 32, lane);
      acc = wmma_bf16(qa[i], b, acc);
    }
    float bval = bias[nb + m];
    size_t base = (size_t)(rowBase + 8 * half) * DDIM + nb + m;
    if (outf) {
#pragma unroll
      for (int v = 0; v < 8; ++v) outf[base + (size_t)v * DDIM] = acc[v] + bval;
    }
    if (outbf) {
#pragma unroll
      for (int v = 0; v < 8; ++v) outbf[base + (size_t)v * DDIM] = f2bf(acc[v] + bval);
    }
  }
}

// ---------------------------------------------------------------- phase 2: sim + argmax
// Key tiles double-buffered in LDS via CDNA5 async loads (ASYNCcnt), overlapped with WMMA.
__global__ __launch_bounds__(256)
void sim_argmax_kernel(const uint16_t* __restrict__ Kall,  // [T,256] bf16
                       const uint16_t* __restrict__ keys,  // [N,256] bf16
                       int* __restrict__ best, int* __restrict__ novel) {
  __shared__ uint32_t kt[2][16 * 128];   // 2 x (16 key rows x 256 bf16) = 16 KB
  const int tid = threadIdx.x;
  const int lane = tid & 31;
  const int wave = tid >> 5;
  const int rowBase = blockIdx.x * 128 + wave * 16;
  const int m = lane & 15, half = lane >> 4;

  BF16Frag qa[8];
#pragma unroll
  for (int i = 0; i < 8; ++i) load_a(qa[i], Kall, DDIM, rowBase, i * 32, lane);

  float bv_[8]; int bn_[8];
#pragma unroll
  for (int v = 0; v < 8; ++v) { bv_[v] = -3e38f; bn_[v] = 0; }

  auto stage = [&](int buf, int nb) {
    const uint32_t* g = (const uint32_t*)keys + (size_t)nb * 128;
#pragma unroll
    for (int i = 0; i < 2; ++i) {                 // 512 b128 per tile / 256 threads
      int idx = tid * 2 + i;
      async_copy_b128(&kt[buf][idx * 4], g + idx * 4);
    }
  };

  const int NC = NSLOT / 16;
  stage(0, 0);
  for (int c = 0; c < NC; ++c) {
    int nb = c * 16;
    __syncthreads();                              // previous buffer free for refill
    if (c + 1 < NC) { stage((c + 1) & 1, nb + 16); ASYNC_WAIT(2); }
    else           { ASYNC_WAIT(0); }
    __syncthreads();                              // chunk c resident in LDS

    v8f acc = {0.f, 0.f, 0.f, 0.f, 0.f, 0.f, 0.f, 0.f};
#pragma unroll
    for (int i = 0; i < 8; ++i) {
      BF16Frag b;
      load_b(b, (const uint16_t*)kt[c & 1], DDIM, 0, i * 32, lane);
      acc = wmma_bf16(qa[i], b, acc);
    }
    int n = nb + m;
#pragma unroll
    for (int v = 0; v < 8; ++v) {
      float val = acc[v] * SCALE;
      if (val > bv_[v]) { bv_[v] = val; bn_[v] = n; }   // strict > keeps first index
    }
  }
  // cross-lane argmax within each 16-lane half (row striping of C frag)
#pragma unroll
  for (int v = 0; v < 8; ++v) {
#pragma unroll
    for (int msk = 1; msk <= 8; msk <<= 1) {
      float ov = __shfl_xor(bv_[v], msk, 32);
      int   on = __shfl_xor(bn_[v], msk, 32);
      if (ov > bv_[v] || (ov == bv_[v] && on < bn_[v])) { bv_[v] = ov; bn_[v] = on; }
    }
  }
  if (m == 0) {
#pragma unroll
    for (int v = 0; v < 8; ++v) {
      int r = rowBase + v + 8 * half;
      best[r]  = bn_[v];
      novel[r] = (bv_[v] < 0.5f) ? 1 : 0;
    }
  }
}

// ---------------------------------------------------------------- phase 3: serial scan
__global__ __launch_bounds__(256)
void scan_kernel(const float* __restrict__ kall, const float* __restrict__ vall,
                 const int* __restrict__ best, const int* __restrict__ novel,
                 const unsigned char* __restrict__ wmask,
                 const float* __restrict__ act0,
                 float* __restrict__ kb, float* __restrict__ vb,
                 float* __restrict__ act_out) {
  __shared__ float acts[NSLOT];
  __shared__ float rval[256];
  __shared__ int   ridx[256];
  const int tid = threadIdx.x;
  for (int i = tid; i < NSLOT; i += 256) acts[i] = act0[i];
  __syncthreads();

  for (int t = 0; t < T_TOK; ++t) {
    int nv = novel[t];                    // uniform across block
    int slot;
    if (nv) {
      float mv = 3e38f; int mi = 0;
      for (int i = 0; i < NSLOT / 256; ++i) {
        int n = tid * (NSLOT / 256) + i;
        float a = acts[n];
        if (a < mv) { mv = a; mi = n; }   // strict < keeps first index
      }
      rval[tid] = mv; ridx[tid] = mi;
      __syncthreads();
      for (int s = 128; s > 0; s >>= 1) {
        if (tid < s) {
          float o = rval[tid + s]; int oi = ridx[tid + s];
          if (o < rval[tid] || (o == rval[tid] && oi < ridx[tid])) { rval[tid] = o; ridx[tid] = oi; }
        }
        __syncthreads();
      }
      slot = ridx[0];
    } else {
      slot = best[t];
    }
    if (wmask[t]) {
      float alpha = nv ? 0.9f : 0.3f;
      float k = kall[(size_t)t * DDIM + tid];
      float v = vall[(size_t)t * DDIM + tid];
      float* kp = kb + (size_t)slot * DDIM + tid;
      float* vp = vb + (size_t)slot * DDIM + tid;
      *kp = (1.f - alpha) * (*kp) + alpha * k;
      *vp = (1.f - alpha) * (*vp) + alpha * v;
      if (tid == 0) acts[slot] = fminf(1.f, acts[slot] + 0.1f);
    }
    __syncthreads();
  }
  for (int i = tid; i < NSLOT; i += 256) act_out[i] = acts[i];
}

// ---------------------------------------------------------------- phase 4: flash read
// K tile (32x256) and V^T slice (256x32) double-buffered via the Tensor Data Mover.
__global__ __launch_bounds__(256)
void attn_kernel(const uint16_t* __restrict__ Qbf,   // [T,256] bf16
                 const uint16_t* __restrict__ Kbbf,  // [N,256] bf16
                 const uint16_t* __restrict__ VbT,   // [256,N] bf16
                 const float* __restrict__ bias,     // [N]
                 float* __restrict__ out) {          // [T,256] f32
  __shared__ uint32_t ktile[2][32 * 128];   // 2 x 16 KB
  __shared__ uint32_t vtile[2][256 * 16];   // 2 x 16 KB
  __shared__ uint16_t ptile[8][16 * 32];    // per-wave P tile (8 KB)
  const int tid = threadIdx.x;
  const int lane = tid & 31;
  const int wave = tid >> 5;
  const int rowBase = blockIdx.x * 128 + wave * 16;
  const int m = lane & 15, half = lane >> 4;

  BF16Frag qa[8];
#pragma unroll
  for (int i = 0; i < 8; ++i) load_a(qa[i], Qbf, DDIM, rowBase, i * 32, lane);

  v8f o[16];
#pragma unroll
  for (int f = 0; f < 16; ++f) o[f] = (v8f){0.f, 0.f, 0.f, 0.f, 0.f, 0.f, 0.f, 0.f};
  float mrun[8], lrun[8];
#pragma unroll
  for (int v = 0; v < 8; ++v) { mrun[v] = -3e38f; lrun[v] = 0.f; }

  auto stage = [&](int buf, int nb) {
#if HAVE_TDM
    if (wave == 0) {
      tdm_load_2d(&ktile[buf][0], Kbbf + (size_t)nb * DDIM, DDIM, 32, DDIM); // 32 key rows
      tdm_load_2d(&vtile[buf][0], VbT + nb, 32, DDIM, NSLOT);                // 256 x 32 slice
    }
#else
    const uint32_t* kg = (const uint32_t*)Kbbf + (size_t)nb * 128;
    const uint32_t* vg = (const uint32_t*)VbT;
#pragma unroll
    for (int i = 0; i < 16; ++i) ktile[buf][tid * 16 + i] = kg[tid * 16 + i];
#pragma unroll
    for (int i = 0; i < 16; ++i) {
      int idx = tid * 16 + i;
      int r = idx >> 4, c2 = idx & 15;
      vtile[buf][idx] = vg[(size_t)r * (NSLOT / 2) + (nb >> 1) + c2];
    }
#endif
  };

  const int NC = NSLOT / 32;
  stage(0, 0);
  for (int c = 0; c < NC; ++c) {
    int nb = c * 32;
    __syncthreads();                       // previous buffer's readers done
    if (c + 1 < NC) stage((c + 1) & 1, nb + 32);
#if HAVE_TDM
    if (wave == 0) {
      if (c + 1 < NC) __builtin_amdgcn_s_wait_tensorcnt(2);  // TDM retires in order
      else            __builtin_amdgcn_s_wait_tensorcnt(0);
    }
#endif
    __syncthreads();                       // chunk c resident in LDS
    const uint16_t* kt = (const uint16_t*)ktile[c & 1];
    const uint16_t* vt = (const uint16_t*)vtile[c & 1];

    // S = Q @ Ktile^T  (two 16x16 column sub-tiles)
    v8f s0 = {0.f, 0.f, 0.f, 0.f, 0.f, 0.f, 0.f, 0.f};
    v8f s1 = {0.f, 0.f, 0.f, 0.f, 0.f, 0.f, 0.f, 0.f};
#pragma unroll
    for (int i = 0; i < 8; ++i) {
      BF16Frag b0, b1;
      load_b(b0, kt, DDIM, 0,  i * 32, lane);
      load_b(b1, kt, DDIM, 16, i * 32, lane);
      s0 = wmma_bf16(qa[i], b0, s0);
      s1 = wmma_bf16(qa[i], b1, s1);
    }
    float bias0 = bias[nb + m];
    float bias1 = bias[nb + 16 + m];

    // online softmax per row; row stats live across each 16-lane half
#pragma unroll
    for (int v = 0; v < 8; ++v) {
      float a0 = s0[v] * SCALE + bias0;
      float a1 = s1[v] * SCALE + bias1;
      float cmax = fmaxf(a0, a1);
#pragma unroll
      for (int msk = 1; msk <= 8; msk <<= 1) cmax = fmaxf(cmax, __shfl_xor(cmax, msk, 32));
      float mnew = fmaxf(mrun[v], cmax);
      float sc = __expf(mrun[v] - mnew);
      float e0 = __expf(a0 - mnew);
      float e1 = __expf(a1 - mnew);
      float ps = e0 + e1;
#pragma unroll
      for (int msk = 1; msk <= 8; msk <<= 1) ps += __shfl_xor(ps, msk, 32);
      lrun[v] = lrun[v] * sc + ps;
      mrun[v] = mnew;
#pragma unroll
      for (int f = 0; f < 16; ++f) o[f][v] *= sc;
      // C-layout -> LDS (wave-private; same-wave LDS ops are in-order, no barrier)
      ptile[wave][(v + 8 * half) * 32 + m]      = f2bf(e0);
      ptile[wave][(v + 8 * half) * 32 + 16 + m] = f2bf(e1);
    }

    // O += P @ V  (A-frag re-read of P, K=32 covers the chunk)
    BF16Frag pa;
    load_a(pa, &ptile[wave][0], 32, 0, 0, lane);
#pragma unroll
    for (int f = 0; f < 16; ++f) {
      BF16Frag vb_;
      load_b(vb_, vt, 32, f * 16, 0, lane);
      o[f] = wmma_bf16(pa, vb_, o[f]);
    }
  }

#pragma unroll
  for (int f = 0; f < 16; ++f) {
#pragma unroll
    for (int v = 0; v < 8; ++v) {
      float l = lrun[v];
      float val = (l > 0.f) ? o[f][v] / l : 0.f;
      out[(size_t)(rowBase + v + 8 * half) * DDIM + f * 16 + m] = val;
    }
  }
}

// ---------------------------------------------------------------- host
extern "C" void kernel_launch(void* const* d_in, const int* in_sizes, int n_in,
                              void* d_out, int out_size, void* d_ws, size_t ws_size,
                              hipStream_t stream) {
  const float* x       = (const float*)d_in[0];
  const unsigned char* wmask = (const unsigned char*)d_in[1]; // bool_
  const float* keys0   = (const float*)d_in[2];
  const float* values0 = (const float*)d_in[3];
  const float* act0    = (const float*)d_in[4];
  const float* Wk      = (const float*)d_in[5];
  const float* bk      = (const float*)d_in[6];
  const float* Wv      = (const float*)d_in[7];
  const float* bv      = (const float*)d_in[8];
  const float* Wq      = (const float*)d_in[9];
  const float* bq      = (const float*)d_in[10];
  float* out = (float*)d_out;

  size_t off = 0;
  auto wsalloc = [&](size_t bytes) -> void* {
    void* p = (char*)d_ws + off;
    off = (off + bytes + 255) & ~(size_t)255;
    return p;
  };
  uint16_t* x_bf     = (uint16_t*)wsalloc((size_t)T_TOK * DDIM * 2);
  uint16_t* wk_bf    = (uint16_t*)wsalloc((size_t)DDIM * DDIM * 2);
  uint16_t* wv_bf    = (uint16_t*)wsalloc((size_t)DDIM * DDIM * 2);
  uint16_t* wq_bf    = (uint16_t*)wsalloc((size_t)DDIM * DDIM * 2);
  uint16_t* keys0_bf = (uint16_t*)wsalloc((size_t)NSLOT * DDIM * 2);
  float*    kall_f   = (float*)   wsalloc((size_t)T_TOK * DDIM * 4);
  float*    vall_f   = (float*)   wsalloc((size_t)T_TOK * DDIM * 4);
  uint16_t* kall_bf  = (uint16_t*)wsalloc((size_t)T_TOK * DDIM * 2);
  uint16_t* q_bf     = (uint16_t*)wsalloc((size_t)T_TOK * DDIM * 2);
  int*      best     = (int*)     wsalloc((size_t)T_TOK * 4);
  int*      novel    = (int*)     wsalloc((size_t)T_TOK * 4);
  float*    kbuf     = (float*)   wsalloc((size_t)NSLOT * DDIM * 4);
  float*    vbuf     = (float*)   wsalloc((size_t)NSLOT * DDIM * 4);
  float*    act_ws   = (float*)   wsalloc((size_t)NSLOT * 4);
  uint16_t* kb_bf    = (uint16_t*)wsalloc((size_t)NSLOT * DDIM * 2);
  uint16_t* vbt_bf   = (uint16_t*)wsalloc((size_t)NSLOT * DDIM * 2);
  float*    bias_ws  = (float*)   wsalloc((size_t)NSLOT * 4);

  const int TB = 256;
  // bf16 conversions of GEMM operands
  cvt_kernel<<<(T_TOK * DDIM + TB - 1) / TB, TB, 0, stream>>>(x, x_bf, T_TOK * DDIM);
  cvt_kernel<<<(DDIM * DDIM + TB - 1) / TB, TB, 0, stream>>>(Wk, wk_bf, DDIM * DDIM);
  cvt_kernel<<<(DDIM * DDIM + TB - 1) / TB, TB, 0, stream>>>(Wv, wv_bf, DDIM * DDIM);
  cvt_kernel<<<(DDIM * DDIM + TB - 1) / TB, TB, 0, stream>>>(Wq, wq_bf, DDIM * DDIM);
  cvt_kernel<<<(NSLOT * DDIM + TB - 1) / TB, TB, 0, stream>>>(keys0, keys0_bf, NSLOT * DDIM);

  // phase 1: k/v/q projections (WMMA, W staged by TDM)
  proj_gemm_kernel<<<T_TOK / 128, TB, 0, stream>>>(x_bf, wk_bf, bk, kall_f, kall_bf);
  proj_gemm_kernel<<<T_TOK / 128, TB, 0, stream>>>(x_bf, wv_bf, bv, vall_f, nullptr);
  proj_gemm_kernel<<<T_TOK / 128, TB, 0, stream>>>(x_bf, wq_bf, bq, nullptr, q_bf);

  // phase 2: similarity vs initial keys + fused argmax/novelty (WMMA + async-to-LDS)
  sim_argmax_kernel<<<T_TOK / 128, TB, 0, stream>>>(kall_bf, keys0_bf, best, novel);

  // phase 3: buffer init + serial scan
  initbuf_kernel<<<(NSLOT * DDIM + TB - 1) / TB, TB, 0, stream>>>(keys0, values0, kbuf, vbuf, NSLOT * DDIM);
  scan_kernel<<<1, TB, 0, stream>>>(kall_f, vall_f, best, novel, wmask, act0, kbuf, vbuf, act_ws);

  // phase 3.5: bf16 buffers (kb row-major, vb transposed) + log-activation bias
  convert2_kernel<<<(NSLOT * DDIM + TB - 1) / TB, TB, 0, stream>>>(kbuf, vbuf, act_ws, kb_bf, vbt_bf, bias_ws);

  // phase 4: content read = flash attention over 4096 slots (WMMA + TDM double-buffer)
  attn_kernel<<<T_TOK / 128, TB, 0, stream>>>(q_bf, kb_bf, vbt_bf, bias_ws, out);
}
